// GIN_43173011259653
// MI455X (gfx1250) — matrix operand order repaired and checked by reading, not compiled, not run on hardware
//
#include <hip/hip_runtime.h>
#include <math.h>

#define N_NODES  50000
#define N_EDGES  800000
#define N_GRAPHS 512
#define D        128
#define NLAYERS  4
#define BN_EPS   1e-5f
#define WPITCH   136   // halves; 272B row pitch -> 16B aligned frags, banks spread 4 apart

typedef __attribute__((ext_vector_type(16))) _Float16 v16h;
typedef __attribute__((ext_vector_type(8)))  _Float16 v8h;
typedef __attribute__((ext_vector_type(8)))  float    v8f;
typedef __attribute__((ext_vector_type(4)))  float    v4f;
typedef __attribute__((ext_vector_type(4)))  int      v4i;

// address-space-qualified pointee types for the async-to-LDS builtin
typedef __attribute__((address_space(1))) v4i g_v4i;   // global <4 x i32>
typedef __attribute__((address_space(3))) v4i l_v4i;   // LDS    <4 x i32>

#if defined(__gfx1250__) && \
    __has_builtin(__builtin_amdgcn_global_load_async_to_lds_b128) && \
    __has_builtin(__builtin_amdgcn_s_wait_asynccnt)
#define USE_ASYNC_LDS 1
#else
#define USE_ASYNC_LDS 0
#endif

// Copy one 16-byte chunk (8 halves) global -> LDS, async when available.
__device__ __forceinline__ void ldsCopy16(const _Float16* g, _Float16* l) {
#if USE_ASYNC_LDS
  __builtin_amdgcn_global_load_async_to_lds_b128(
      (g_v4i*)(unsigned long long)(const void*)g,
      (l_v4i*)(unsigned int)(unsigned long long)(const void*)l,
      0, 0);
#else
  *(v8h*)l = *(const v8h*)g;
#endif
}

__device__ __forceinline__ void ldsCopyWait() {
#if USE_ASYNC_LDS
  __builtin_amdgcn_s_wait_asynccnt(0);
#endif
}

// ---------------------------------------------------------------- utilities
__global__ void k_fill(float* __restrict__ p, long n, float val) {
  long i = (long)blockIdx.x * blockDim.x + threadIdx.x;
  long stride = (long)gridDim.x * blockDim.x;
  for (; i < n; i += stride) p[i] = val;
}

__device__ __forceinline__ void atomicMaxF(float* addr, float v) {
  if (v >= 0.0f) atomicMax((int*)addr, __float_as_int(v));
  else           atomicMin((unsigned int*)addr, __float_as_uint(v));
}

// --------------------------------- one-time weight prep: f32 -> f16, transposed
// Wt[mat][n][k] = (half) W[mat][k][n]   (D*D = 16384 = 1<<14)
__global__ __launch_bounds__(256)
void k_prep(const float* __restrict__ W, _Float16* __restrict__ Wt, int nMat) {
  const int idx = blockIdx.x * 256 + threadIdx.x;
  if (idx >= nMat * D * D) return;
  const int mat = idx >> 14;
  const int r   = idx & (D * D - 1);
  const int k   = r >> 7, n = r & 127;
  Wt[(long)mat * D * D + n * D + k] = (_Float16)W[(long)mat * D * D + k * D + n];
}

// -------------------------------------------- one-time CSR build (graph static)
__global__ __launch_bounds__(256)
void k_hist(const int* __restrict__ dst, int* __restrict__ deg, int nEdges) {
  const int e = blockIdx.x * 256 + threadIdx.x;
  if (e < nEdges) atomicAdd(&deg[dst[e]], 1);
}

// single-block exclusive scan over 50000 degrees -> rowStart, cursor
__global__ __launch_bounds__(256)
void k_scan(const int* __restrict__ deg, int* __restrict__ rowStart,
            int* __restrict__ cursor) {
  __shared__ int part[256];
  const int t = threadIdx.x;
  const int chunk = (N_NODES + 255) / 256;            // 196
  const int lo = t * chunk;
  const int hi = (lo + chunk < N_NODES) ? lo + chunk : N_NODES;
  int s = 0;
  for (int i = lo; i < hi; ++i) s += deg[i];
  part[t] = s;
  __syncthreads();
  for (int off = 1; off < 256; off <<= 1) {           // Hillis-Steele inclusive
    const int v = (t >= off) ? part[t - off] : 0;
    __syncthreads();
    part[t] += v;
    __syncthreads();
  }
  int base = (t == 0) ? 0 : part[t - 1];
  for (int i = lo; i < hi; ++i) {
    rowStart[i] = base;
    cursor[i]   = base;
    base += deg[i];
  }
  if (t == 255) rowStart[N_NODES] = base;             // == N_EDGES
}

__global__ __launch_bounds__(256)
void k_csr(const int* __restrict__ src, const int* __restrict__ dst,
           int* __restrict__ cursor, int* __restrict__ csrSrc, int nEdges) {
  const int e = blockIdx.x * 256 + threadIdx.x;
  if (e >= nEdges) return;
  const int pos = atomicAdd(&cursor[dst[e]], 1);
  csrSrc[pos] = src[e];
}

// ------------------------- atomic-free aggregation: z[n] = h[n] + sum h[nbr]
// one wave per node; lane owns 4 features; gathers are L2-resident.
__global__ __launch_bounds__(256)
void k_gather(const float* __restrict__ h, const int* __restrict__ rowStart,
              const int* __restrict__ csrSrc, float* __restrict__ z) {
  const int node = blockIdx.x * 8 + (threadIdx.x >> 5);
  const int lane = threadIdx.x & 31;
  if (node >= N_NODES) return;
  const int f0 = lane * 4;
  v4f acc = *(const v4f*)(h + (long)node * D + f0);   // self term (eps = 0)
  const int e0 = rowStart[node], e1 = rowStart[node + 1];
  for (int j = e0; j < e1; ++j) {
    const int s = csrSrc[j];                          // broadcast load, coalesces
    const v4f v = *(const v4f*)(h + (long)s * D + f0);
    acc[0] += v[0]; acc[1] += v[1]; acc[2] += v[2]; acc[3] += v[3];
  }
  *(v4f*)(z + (long)node * D + f0) = acc;
}

// --------------------------------------------- fused WMMA GEMM + BN stats
// mode 0: A = X                 (identity)
// mode 1: A = relu(X*sc + sh)   (BN+ReLU prologue, per input column)
// Wt: pre-transposed f16 weights [n][k]. Epilogue: Y = A @ W, column sum/sumsq.
__global__ __launch_bounds__(256)
void k_gemm(const float* __restrict__ X,
            const float* __restrict__ scsh, const _Float16* __restrict__ Wt,
            float* __restrict__ Y, float* __restrict__ stats,
            int nRows, int mode) {
  __shared__ _Float16 sW[D * WPITCH];
  __shared__ float    sScsh[2 * D];
  __shared__ float    sStats[2 * D];

  const int t = threadIdx.x;
  // stage Wt (128x128 halves) into padded LDS: 2048 x 16B async chunks
  #pragma unroll
  for (int i = 0; i < 8; ++i) {
    const int c16 = t + i * 256;
    const int n = c16 >> 4, c = (c16 & 15) * 8;
    ldsCopy16(Wt + n * D + c, &sW[n * WPITCH + c]);
  }
  sStats[t] = 0.0f;                       // blockDim == 2*D == 256
  if (mode == 1) sScsh[t] = scsh[t];
  ldsCopyWait();
  __syncthreads();

  const int wave    = t >> 5;
  const int lane    = t & 31;
  const int m       = lane & 15;
  const bool hiHalf = lane >= 16;
  const int rowBase = blockIdx.x * 128 + wave * 16;
  const int row     = rowBase + m;        // A-matrix row this lane feeds

  v8f acc[8];
  #pragma unroll
  for (int nt = 0; nt < 8; ++nt)
    #pragma unroll
    for (int r = 0; r < 8; ++r) acc[nt][r] = 0.0f;

  #pragma unroll
  for (int kt = 0; kt < 4; ++kt) {
    // ---- A fragment (16x32 f16): lanes 0-15 K {0..7,16..23}, lanes 16-31 +8
    v16h a;
    const int k0 = kt * 32 + (hiHalf ? 8 : 0);
    if (row < nRows) {
      const float* xp = X + (long)row * D + k0;
      float vals[16];
      #pragma unroll
      for (int j = 0; j < 8; ++j) { vals[j] = xp[j]; vals[8 + j] = xp[16 + j]; }
      if (mode == 1) {
        #pragma unroll
        for (int j = 0; j < 8; ++j) {
          const int ka = k0 + j, kb = k0 + 16 + j;
          vals[j]     = fmaxf(vals[j]     * sScsh[ka] + sScsh[D + ka], 0.0f);
          vals[8 + j] = fmaxf(vals[8 + j] * sScsh[kb] + sScsh[D + kb], 0.0f);
        }
      }
      #pragma unroll
      for (int j = 0; j < 16; ++j) a[j] = (_Float16)vals[j];
    } else {
      #pragma unroll
      for (int j = 0; j < 16; ++j) a[j] = (_Float16)0.0f;
    }

    // ---- 8 N-tiles: B fragment from LDS, accumulate
    #pragma unroll
    for (int nt = 0; nt < 8; ++nt) {
      const int n  = nt * 16 + m;
      const int kb = kt * 32 + (hiHalf ? 8 : 0);
      const _Float16* bp = &sW[n * WPITCH + kb];
      const v8h blo = *(const v8h*)bp;
      const v8h bhi = *(const v8h*)(bp + 16);
      v16h b;
      #pragma unroll
      for (int j = 0; j < 8; ++j) { b[j] = blo[j]; b[8 + j] = bhi[j]; }
      acc[nt] = __builtin_amdgcn_wmma_f32_16x16x32_f16(
          false, a, false, b, (short)0, acc[nt], false, false);
    }
  }

  // ---- epilogue: store + column sum/sumsq (LDS reduce, then 1 global atomic)
  #pragma unroll
  for (int nt = 0; nt < 8; ++nt) {
    const int col = nt * 16 + m;
    float s = 0.0f, s2 = 0.0f;
    #pragma unroll
    for (int r = 0; r < 8; ++r) {
      const int rr = rowBase + r + (hiHalf ? 8 : 0);
      if (rr < nRows) {
        const float v = acc[nt][r];
        Y[(long)rr * D + col] = v;
        s += v; s2 += v * v;
      }
    }
    atomicAdd(&sStats[col], s);
    atomicAdd(&sStats[D + col], s2);
  }
  __syncthreads();
  atomicAdd(&stats[t], sStats[t]);
}

// ----------------------------------------------- BN finalize -> scale/shift
__global__ void k_bnfin(const float* __restrict__ stats,
                        const float* __restrict__ gamma,
                        const float* __restrict__ beta,
                        float* __restrict__ scsh) {
  const int c = threadIdx.x;
  const float inv_n = 1.0f / (float)N_NODES;
  const float mean = stats[c] * inv_n;
  const float var  = stats[D + c] * inv_n - mean * mean;
  const float sc   = gamma[c] * rsqrtf(var + BN_EPS);
  scsh[c]     = sc;
  scsh[D + c] = beta[c] - mean * sc;
}

// --------------------------- BN+ReLU apply, write h, fused segment-max pool
__global__ __launch_bounds__(256)
void k_act_pool(const float* __restrict__ Z, const float* __restrict__ scsh,
                float* __restrict__ hOut, float* __restrict__ pooled,
                const int* __restrict__ gid, int applyBn) {
  const long idx = (long)blockIdx.x * blockDim.x + threadIdx.x;
  const int node = (int)(idx >> 5);
  if (node >= N_NODES) return;
  const int f0 = (int)(idx & 31) * 4;
  v4f v = *(const v4f*)(Z + (long)node * D + f0);
  if (applyBn) {
    #pragma unroll
    for (int j = 0; j < 4; ++j)
      v[j] = fmaxf(v[j] * scsh[f0 + j] + scsh[D + f0 + j], 0.0f);
  }
  *(v4f*)(hOut + (long)node * D + f0) = v;
  const int g = gid[node];
  float* pp = pooled + (long)g * D + f0;
  #pragma unroll
  for (int j = 0; j < 4; ++j) atomicMaxF(pp + j, v[j]);
}

// -------------------------------------- readout: sum_i pooled_i @ W_i + b_i
__global__ __launch_bounds__(256)
void k_readout(const float* __restrict__ pooled, const _Float16* __restrict__ WtP,
               const float* __restrict__ predb, float* __restrict__ out) {
  __shared__ _Float16 sW[D * WPITCH];
  const int t = threadIdx.x;
  const int wave = t >> 5, lane = t & 31, m = lane & 15;
  const bool hiHalf = lane >= 16;
  const int rowBase = blockIdx.x * 128 + wave * 16;   // graph rows, always < 512
  const int row = rowBase + m;

  v8f acc[8];
  float bias[8];
  #pragma unroll
  for (int nt = 0; nt < 8; ++nt) {
    bias[nt] = 0.0f;
    #pragma unroll
    for (int r = 0; r < 8; ++r) acc[nt][r] = 0.0f;
  }

  for (int layer = 0; layer <= NLAYERS; ++layer) {
    __syncthreads();
    const _Float16* Wt = WtP + (long)layer * D * D;
    #pragma unroll
    for (int i = 0; i < 8; ++i) {
      const int c16 = t + i * 256;
      const int n = c16 >> 4, c = (c16 & 15) * 8;
      ldsCopy16(Wt + n * D + c, &sW[n * WPITCH + c]);
    }
    ldsCopyWait();
    __syncthreads();

    const float* P = pooled + (long)layer * N_GRAPHS * D;
    #pragma unroll
    for (int kt = 0; kt < 4; ++kt) {
      v16h a;
      const int k0 = kt * 32 + (hiHalf ? 8 : 0);
      const float* xp = P + (long)row * D + k0;
      #pragma unroll
      for (int j = 0; j < 8; ++j) {
        a[j]     = (_Float16)xp[j];
        a[8 + j] = (_Float16)xp[16 + j];
      }
      #pragma unroll
      for (int nt = 0; nt < 8; ++nt) {
        const int n  = nt * 16 + m;
        const int kb = kt * 32 + (hiHalf ? 8 : 0);
        const _Float16* bp = &sW[n * WPITCH + kb];
        const v8h blo = *(const v8h*)bp;
        const v8h bhi = *(const v8h*)(bp + 16);
        v16h b;
        #pragma unroll
        for (int j = 0; j < 8; ++j) { b[j] = blo[j]; b[8 + j] = bhi[j]; }
        acc[nt] = __builtin_amdgcn_wmma_f32_16x16x32_f16(
            false, a, false, b, (short)0, acc[nt], false, false);
      }
    }
    #pragma unroll
    for (int nt = 0; nt < 8; ++nt) bias[nt] += predb[layer * D + nt * 16 + m];
  }

  #pragma unroll
  for (int nt = 0; nt < 8; ++nt) {
    const int col = nt * 16 + m;
    #pragma unroll
    for (int r = 0; r < 8; ++r) {
      const int rr = rowBase + r + (hiHalf ? 8 : 0);
      out[(long)rr * D + col] = acc[nt][r] + bias[nt];
    }
  }
}

// ------------------------------------------------------------------- driver
extern "C" void kernel_launch(void* const* d_in, const int* in_sizes, int n_in,
                              void* d_out, int out_size, void* d_ws, size_t ws_size,
                              hipStream_t stream) {
  const float* h_in   = (const float*)d_in[0];
  const int*   src    = (const int*)d_in[1];
  const int*   dst    = (const int*)d_in[2];
  const int*   gid    = (const int*)d_in[3];
  const float* W1s    = (const float*)d_in[4];
  const float* W2s    = (const float*)d_in[5];
  const float* bn1_g  = (const float*)d_in[6];
  const float* bn1_b  = (const float*)d_in[7];
  const float* bn2_g  = (const float*)d_in[8];
  const float* bn2_b  = (const float*)d_in[9];
  const float* predW  = (const float*)d_in[10];
  const float* predb  = (const float*)d_in[11];
  float* out = (float*)d_out;

  const long ND = (long)N_NODES * D;                 // 6,400,000
  float* ws = (float*)d_ws;
  float* h_cur  = ws;                                // ND
  float* z      = h_cur + ND;                        // ND
  float* y      = z + ND;                            // ND
  float* pooled = y + ND;                            // 5*512*128
  float* stats  = pooled + (long)(NLAYERS + 1) * N_GRAPHS * D;   // 2*256
  float* scsh   = stats + 512;                       // 2*256
  _Float16* wh  = (_Float16*)(scsh + 512);           // 13 * 16384 halves (f16, transposed)
  _Float16* w1h = wh;                                // 4 matrices
  _Float16* w2h = wh + 4L * D * D;                   // 4 matrices
  _Float16* wph = wh + 8L * D * D;                   // 5 matrices
  int* ints     = (int*)(wh + 13L * D * D);          // CSR work
  int* deg      = ints;                              // N_NODES
  int* cursor   = deg + N_NODES;                     // N_NODES
  int* rowStart = cursor + N_NODES;                  // N_NODES + 1
  int* csrSrc   = rowStart + N_NODES + 1;            // N_EDGES
  const size_t needBytes = (size_t)((char*)(csrSrc + N_EDGES) - (char*)ws);
  if (ws_size < needBytes) return;

  const int gGemm   = (N_NODES + 127) / 128;         // 391
  const int gNode   = (int)((ND / 4 + 255) / 256);   // act_pool grid (node*32 threads)
  const int gEdge   = (N_EDGES + 255) / 256;         // 3125
  const int gGather = (N_NODES + 7) / 8;             // 6250

  // one-time weight prep: f32 -> f16, transposed
  k_prep<<<(4 * D * D + 255) / 256, 256, 0, stream>>>(W1s, w1h, 4);
  k_prep<<<(4 * D * D + 255) / 256, 256, 0, stream>>>(W2s, w2h, 4);
  k_prep<<<(5 * D * D + 255) / 256, 256, 0, stream>>>(predW, wph, 5);

  // one-time CSR build (graph is static across layers)
  k_fill<<<64, 256, 0, stream>>>((float*)deg, N_NODES, 0.0f);   // int 0
  k_hist<<<gEdge, 256, 0, stream>>>(dst, deg, N_EDGES);
  k_scan<<<1, 256, 0, stream>>>(deg, rowStart, cursor);
  k_csr<<<gEdge, 256, 0, stream>>>(src, dst, cursor, csrSrc, N_EDGES);

  // pooled[0..4] := -inf ; layer-0 pool from raw h (also copies h -> h_cur)
  k_fill<<<2048, 256, 0, stream>>>(pooled, (long)(NLAYERS + 1) * N_GRAPHS * D, -INFINITY);
  k_act_pool<<<gNode, 256, 0, stream>>>(h_in, scsh, h_cur, pooled, gid, 0);

  for (int i = 0; i < NLAYERS; ++i) {
    _Float16* W1 = w1h + (long)i * D * D;
    _Float16* W2 = w2h + (long)i * D * D;

    // z = h + segment_sum(h[src], dst)  — atomic-free gather
    k_gather<<<gGather, 256, 0, stream>>>(h_cur, rowStart, csrSrc, z);

    // GEMM1: y = z @ W1, accumulate column stats of y
    k_fill<<<1, 256, 0, stream>>>(stats, 512, 0.0f);
    k_gemm<<<gGemm, 256, 0, stream>>>(z, scsh, W1, y, stats, N_NODES, 0);
    k_bnfin<<<1, D, 0, stream>>>(stats, bn1_g + i * D, bn1_b + i * D, scsh);

    // GEMM2: z = relu(BN1(y)) @ W2, accumulate column stats of z
    k_fill<<<1, 256, 0, stream>>>(stats + 256, 512, 0.0f);
    k_gemm<<<gGemm, 256, 0, stream>>>(y, scsh, W2, z, stats + 256, N_NODES, 1);
    k_bnfin<<<1, D, 0, stream>>>(stats + 256, bn2_g + i * D, bn2_b + i * D, scsh + 256);

    // h = relu(BN2(z)); pool into pooled[i+1]
    k_act_pool<<<gNode, 256, 0, stream>>>(z, scsh + 256, h_cur,
                                          pooled + (long)(i + 1) * N_GRAPHS * D, gid, 1);
  }

  k_readout<<<N_GRAPHS / 128, 256, 0, stream>>>(pooled, wph, predb, out);
}
